// Decoder_Attention_32538672235168
// MI455X (gfx1250) — matrix operand order repaired and checked by reading, not compiled, not run on hardware
//
#include <hip/hip_runtime.h>
#include <hip/hip_bf16.h>

#define B_  32
#define T_  64
#define S_  512
#define H_  1024
#define E_  1024

typedef __attribute__((ext_vector_type(16))) __bf16 v16bf;
typedef __attribute__((ext_vector_type(8)))  float  v8f;

union BFrag {
    v16bf v;
    unsigned short s[16];
    uint4 q[2];
};

__device__ __forceinline__ unsigned short f2bf(float f) {
    unsigned int u = __float_as_uint(f);
    unsigned int r = u + 0x7FFFu + ((u >> 16) & 1u);   // round-to-nearest-even
    return (unsigned short)(r >> 16);
}

// A-fragment (16x32 bf16, wave32) inverse map: element (row m of 32, k) ->
// flat offset in packed buffer laid out [kt][mt][lane][e], 512 halves per
// (kt,mt) tile.  Forward map: k = 2r + h + 8g (r<4), k = 2r + 8 + h + 8g (r>=4),
// lane = (m%16) + 16g, e = 2r + h.
__device__ __forceinline__ size_t afrag_off(int m, int k) {
    int kt   = k >> 5;
    int kl   = k & 31;
    int g    = (kl >> 3) & 1;
    int e    = (kl & 7) | ((kl >> 4) << 3);
    int lane = (m & 15) + (g << 4);
    int mt   = m >> 4;
    return (((size_t)(kt * 2 + mt) * 32 + lane) << 4) + e;
}

// ---------------------------------------------------------------------------
// Pack a (virtual) K x N f32 row-major weight matrix into bf16 WMMA B-fragment
// order: tile = nt*Ktiles + kt; within tile lane-major, 16 halves per lane in
// element order e, k = (e%8) + 8*(lane>=16) + 16*(e/8), n = lane%16.
// Columns n >= nsplit come from B1.
// ---------------------------------------------------------------------------
__global__ __launch_bounds__(256)
void pack_b_kernel(const float* __restrict__ B0, int ld0, int nsplit,
                   const float* __restrict__ B1, int ld1,
                   int Ktiles, unsigned short* __restrict__ out, size_t total)
{
    size_t idx = (size_t)blockIdx.x * 256 + threadIdx.x;
    if (idx >= total) return;
    int e    = (int)(idx & 15);
    int lane = (int)((idx >> 4) & 31);
    size_t tile = idx >> 9;
    int kt = (int)(tile % (size_t)Ktiles);
    int nt = (int)(tile / (size_t)Ktiles);
    int g  = lane >> 4;
    int nl = lane & 15;
    int r  = e >> 1, hh = e & 1;
    int k  = kt * 32 + (e & 7) + 8 * g + 16 * (e >> 3);
    int n  = nt * 16 + nl;
    float v = (n < nsplit) ? B0[(size_t)k * ld0 + n]
                           : B1[(size_t)k * ld1 + (n - nsplit)];
    out[idx] = f2bf(v);
}

// Pack a 32 x 1024 f32 activation matrix (row stride ld) into A-fragments.
__global__ __launch_bounds__(256)
void pack_a_kernel(const float* __restrict__ A, int ld,
                   unsigned short* __restrict__ out)
{
    int idx = blockIdx.x * 256 + threadIdx.x;   // 0 .. 32*1024-1
    int m = idx >> 10;
    int k = idx & 1023;
    out[afrag_off(m, k)] = f2bf(A[(size_t)m * ld + k]);
}

// Pack the whole dec_seq (B,T,E) into per-timestep A-fragment blocks:
// out[t * 32768 + afrag_off(b, e)]
__global__ __launch_bounds__(256)
void pack_dec_kernel(const float* __restrict__ dec,
                     unsigned short* __restrict__ out)
{
    int idx = blockIdx.x * 256 + threadIdx.x;   // 0 .. B*T*E-1
    int e = idx & (E_ - 1);
    int t = (idx >> 10) & (T_ - 1);
    int b = idx >> 16;
    out[(size_t)t * 32768 + afrag_off(b, e)] = f2bf(dec[idx]);
}

// ---------------------------------------------------------------------------
// WMMA GEMM: C(32 x N) = A(32 x K) * B(K x N), bf16 fragments, f32 accum.
// A k-tiles: kt < Kt1 from A0pk, kt >= Kt1 from A1pk (both pre-packed).
// Two clean loops -> no per-iteration source select in the hot path.
// One wave: 2 M-tiles x 4 N-tiles; block = 4 waves = 16 N-tiles.
// ---------------------------------------------------------------------------
__global__ __launch_bounds__(128)
void gemm_wmma_kernel(const unsigned short* __restrict__ A0pk,
                      const unsigned short* __restrict__ A1pk,
                      int Kt1, int Ktiles,
                      const unsigned short* __restrict__ Bpk,
                      float* __restrict__ C, int ldc)
{
    const int lane = threadIdx.x & 31;
    const int wave = threadIdx.x >> 5;
    const int m    = lane & 15;
    const int g    = lane >> 4;
    const int nt0  = (blockIdx.x * 4 + wave) * 4;

    v8f acc[4][2] = {};

    auto body = [&](const unsigned short* __restrict__ ap, int kt) {
        BFrag a[2];
#pragma unroll
        for (int mt = 0; mt < 2; ++mt) {
            const uint4* pa = (const uint4*)(ap + ((size_t)mt << 9) + ((size_t)lane << 4));
            a[mt].q[0] = pa[0];
            a[mt].q[1] = pa[1];
        }
#pragma unroll
        for (int w = 0; w < 4; ++w) {
            BFrag b;
            const uint4* p = (const uint4*)(Bpk + (((size_t)(nt0 + w) * Ktiles + kt) << 9)
                                                + ((size_t)lane << 4));
            b.q[0] = p[0];
            b.q[1] = p[1];
            acc[w][0] = __builtin_amdgcn_wmma_f32_16x16x32_bf16(
                false, a[0].v, false, b.v, (short)0, acc[w][0], false, false);
            acc[w][1] = __builtin_amdgcn_wmma_f32_16x16x32_bf16(
                false, a[1].v, false, b.v, (short)0, acc[w][1], false, false);
        }
    };

    for (int kt = 0; kt < Kt1; ++kt)
        body(A0pk + ((size_t)kt << 10), kt);
    for (int kt = Kt1; kt < Ktiles; ++kt)
        body(A1pk + ((size_t)(kt - Kt1) << 10), kt);

#pragma unroll
    for (int w = 0; w < 4; ++w) {
        const int col = (nt0 + w) * 16 + m;
#pragma unroll
        for (int mt = 0; mt < 2; ++mt) {
#pragma unroll
            for (int e = 0; e < 8; ++e) {
                int row = mt * 16 + e + 8 * g;
                C[(size_t)row * ldc + col] = acc[w][mt][e];
            }
        }
    }
}

// ---------------------------------------------------------------------------
// Attention: one block per batch element.
//   score[s] = dot(enc[b,s,:], proj[b,:]); softmax; context = attn @ enc[b]
// proj = columns [0,1024) of hb (32 x 4096). Context written directly as
// packed bf16 A-fragments (ctx_pk).
// ---------------------------------------------------------------------------
__global__ __launch_bounds__(1024)
void attn_kernel(const float* __restrict__ enc,
                 const float* __restrict__ hb,
                 unsigned short* __restrict__ ctx_pk)
{
    __shared__ float pLDS[H_];
    __shared__ float sLDS[S_];
    __shared__ float redA[32];
    __shared__ float redB[32];

    const int b    = blockIdx.x;
    const int tid  = threadIdx.x;
    const int lane = tid & 31;
    const int wave = tid >> 5;
    const float* encb = enc + (size_t)b * S_ * H_;

    pLDS[tid] = hb[(size_t)b * 4096 + tid];
    __syncthreads();

    for (int s = wave; s < S_; s += 32) {
        const float* row = encb + (size_t)s * H_;
        float acc = 0.f;
        for (int h = lane; h < H_; h += 32) acc += row[h] * pLDS[h];
#pragma unroll
        for (int off = 16; off > 0; off >>= 1) acc += __shfl_xor(acc, off, 32);
        if (lane == 0) sLDS[s] = acc;
    }
    __syncthreads();

    float val = (tid < S_) ? sLDS[tid] : -__builtin_inff();
    float mx = val;
#pragma unroll
    for (int off = 16; off > 0; off >>= 1) mx = fmaxf(mx, __shfl_xor(mx, off, 32));
    if (lane == 0) redA[wave] = mx;
    __syncthreads();
    if (wave == 0) {
        float x = redA[lane];
#pragma unroll
        for (int off = 16; off > 0; off >>= 1) x = fmaxf(x, __shfl_xor(x, off, 32));
        if (lane == 0) redA[0] = x;
    }
    __syncthreads();
    mx = redA[0];

    float ex = (tid < S_) ? __expf(val - mx) : 0.f;
    float sm = ex;
#pragma unroll
    for (int off = 16; off > 0; off >>= 1) sm += __shfl_xor(sm, off, 32);
    if (lane == 0) redB[wave] = sm;
    __syncthreads();
    if (wave == 0) {
        float x = redB[lane];
#pragma unroll
        for (int off = 16; off > 0; off >>= 1) x += __shfl_xor(x, off, 32);
        if (lane == 0) redB[0] = x;
    }
    __syncthreads();
    const float inv = 1.f / redB[0];
    if (tid < S_) sLDS[tid] = ex * inv;
    __syncthreads();

    float acc = 0.f;
#pragma unroll 4
    for (int s = 0; s < S_; ++s) acc += sLDS[s] * encb[(size_t)s * H_ + tid];
    ctx_pk[afrag_off(b, tid)] = f2bf(acc);
}

// ---------------------------------------------------------------------------
// GRU gate update; also re-packs h_new into bf16 A-fragments for next step.
// ---------------------------------------------------------------------------
__global__ __launch_bounds__(256)
void gate_kernel(const float* __restrict__ gx, const float* __restrict__ hb,
                 const float* __restrict__ bias,
                 float* __restrict__ h, unsigned short* __restrict__ hA_pk,
                 float* __restrict__ out, int t)
{
    const int idx = blockIdx.x * 256 + threadIdx.x;   // 0 .. 32*1024-1
    const int b = idx >> 10;
    const int n = idx & 1023;
    const float* gxb = gx + (size_t)b * 3072;
    const float* ghb = hb + (size_t)b * 4096 + 1024;

    float xz = gxb[n]        + bias[n];
    float xr = gxb[1024 + n] + bias[1024 + n];
    float xh = gxb[2048 + n] + bias[2048 + n];
    float hz = ghb[n];
    float hr = ghb[1024 + n];
    float hh = ghb[2048 + n];

    float z    = 1.f / (1.f + __expf(-(xz + hz)));
    float r    = 1.f / (1.f + __expf(-(xr + hr)));
    float cand = tanhf(xh + r * hh);
    float hn   = z * h[idx] + (1.f - z) * cand;

    h[idx] = hn;
    hA_pk[afrag_off(b, n)] = f2bf(hn);
    out[((size_t)b * T_ + t) * H_ + n] = hn;
}

// ---------------------------------------------------------------------------
extern "C" void kernel_launch(void* const* d_in, const int* in_sizes, int n_in,
                              void* d_out, int out_size, void* d_ws, size_t ws_size,
                              hipStream_t stream)
{
    const float* dec_seq = (const float*)d_in[0];   // (B,T,E)
    const float* enc     = (const float*)d_in[1];   // (B,S,H)
    const float* last_h  = (const float*)d_in[2];   // (B,H)
    const float* W       = (const float*)d_in[3];   // (H,H)
    const float* kernelW = (const float*)d_in[4];   // (E+H, 3H)
    const float* rker    = (const float*)d_in[5];   // (H, 3H)
    const float* bias    = (const float*)d_in[6];   // (3H,)
    float* out = (float*)d_out;

    char* ws = (char*)d_ws;
    float* h   = (float*)ws;  ws += (size_t)B_ * H_ * 4;            // 128 KB
    float* hb  = (float*)ws;  ws += (size_t)B_ * 4096 * 4;          // 512 KB (proj | gh)
    float* gx  = (float*)ws;  ws += (size_t)B_ * 3072 * 4;          // 384 KB
    unsigned short* hA_pk  = (unsigned short*)ws; ws += 32768 * 2;               // 64 KB
    unsigned short* ctx_pk = (unsigned short*)ws; ws += 32768 * 2;               // 64 KB
    unsigned short* dec_pk = (unsigned short*)ws; ws += (size_t)T_ * 32768 * 2;  // 4 MB
    unsigned short* Wpk    = (unsigned short*)ws; ws += (size_t)1024 * 4096 * 2; // 8 MB
    unsigned short* Kpk    = (unsigned short*)ws; ws += (size_t)2048 * 3072 * 2; // 12 MB

    // One-time (per call) packing
    {
        size_t tot = (size_t)1024 * 4096;   // [W | recurrent] -> Ktiles=32
        pack_b_kernel<<<(unsigned)((tot + 255) / 256), 256, 0, stream>>>(
            W, 1024, 1024, rker, 3072, 32, Wpk, tot);
    }
    {
        size_t tot = (size_t)2048 * 3072;   // kernel -> Ktiles=64
        pack_b_kernel<<<(unsigned)((tot + 255) / 256), 256, 0, stream>>>(
            kernelW, 3072, 3072, kernelW, 3072, 64, Kpk, tot);
    }
    pack_dec_kernel<<<(B_ * T_ * E_) / 256, 256, 0, stream>>>(dec_seq, dec_pk);
    pack_a_kernel<<<(B_ * H_) / 256, 256, 0, stream>>>(last_h, H_, hA_pk);
    hipMemcpyAsync(h, last_h, (size_t)B_ * H_ * 4, hipMemcpyDeviceToDevice, stream);

    for (int t = 0; t < T_; ++t) {
        // hb = h @ [W | recurrent]   (N = 4096 -> 16 blocks)
        gemm_wmma_kernel<<<16, 128, 0, stream>>>(hA_pk, hA_pk, 32, 32, Wpk, hb, 4096);
        // attention: score, softmax, context (packed)
        attn_kernel<<<B_, 1024, 0, stream>>>(enc, hb, ctx_pk);
        // gx = [x_t | context] @ kernel   (N = 3072 -> 12 blocks)
        gemm_wmma_kernel<<<12, 128, 0, stream>>>(dec_pk + (size_t)t * 32768, ctx_pk,
                                                 32, 64, Kpk, gx, 3072);
        // gates + hidden update + output + h re-pack
        gate_kernel<<<(B_ * H_) / 256, 256, 0, stream>>>(gx, hb, bias, h, hA_pk, out, t);
    }
}
